// TrueEvolvingAttention_53927609368880
// MI455X (gfx1250) — matrix operand order repaired and checked by reading, not compiled, not run on hardware
//
#include <hip/hip_runtime.h>
#include <math.h>

// ---------------------------------------------------------------------------
// TrueEvolvingAttention for MI455X (gfx1250, wave32, WMMA bf16 16x16x32)
// B=2, T=2048, D_MODEL=1024, H=16, d=64
// ---------------------------------------------------------------------------

#define B_   2
#define T_   2048
#define DM_  1024
#define H_   16
#define HD_  64
#define BH_  (B_ * H_)

typedef __attribute__((ext_vector_type(16))) __bf16 v16bf;
typedef __attribute__((ext_vector_type(8)))  __bf16 v8bf;
typedef __attribute__((ext_vector_type(8)))  float  v8f;

// ---------------- helpers ----------------

__device__ __forceinline__ unsigned short f2bf(float f) {
    union { float f; unsigned u; } x; x.f = f;
    unsigned r = x.u + 0x7FFFu + ((x.u >> 16) & 1u);
    return (unsigned short)(r >> 16);
}

__device__ __forceinline__ v16bf cat16(v8bf lo, v8bf hi) {
    return __builtin_shufflevector(lo, hi, 0,1,2,3,4,5,6,7,8,9,10,11,12,13,14,15);
}

// A-fragment (16x32 bf16, M x K): lanes 0-15 hold row M=lane, K {0..7,16..23};
// lanes 16-31 hold row M=lane-16, K {8..15,24..31}. Source row-major, ld elems.
__device__ __forceinline__ v16bf load_a_frag(const unsigned short* base, int ld) {
    int l = threadIdx.x & 31;
    int row = l & 15;
    int hh  = l >> 4;
    const unsigned short* p = base + row * ld + hh * 8;
    v8bf lo = *(const v8bf*)(p);
    v8bf hi = *(const v8bf*)(p + 16);
    return cat16(lo, hi);
}

// B-fragment (32x16 bf16, K x N) with B[k][n] = X[n][k], X row-major ld elems:
// lanes 0-15 hold col N=lane, K {0..7,16..23}; lanes 16-31 hold col N=lane-16, K {8..15,24..31}.
__device__ __forceinline__ v16bf load_b_frag(const unsigned short* base, int ld) {
    int l = threadIdx.x & 31;
    int n  = l & 15;
    int hh = l >> 4;
    const unsigned short* p = base + n * ld + hh * 8;
    v8bf lo = *(const v8bf*)(p);
    v8bf hi = *(const v8bf*)(p + 16);
    return cat16(lo, hi);
}

__device__ __forceinline__ v8f wmma_bf16(v16bf a, v16bf b, v8f c) {
    return __builtin_amdgcn_wmma_f32_16x16x32_bf16(false, a, false, b, (short)0, c, false, false);
}

// ---------------- kernel 0: f32 -> bf16 convert ----------------

__global__ void k_conv_bf16(const float* __restrict__ in,
                            unsigned short* __restrict__ out, int n) {
    int i = blockIdx.x * blockDim.x + threadIdx.x;
    if (i < n) out[i] = f2bf(in[i]);
}

// ---------------- kernel 1: GEMM  O[m,n] = sum_k A[m,k] * W[n,k] ----------------
// A: [M,K] bf16 row-major, W: [N,K] bf16 row-major, O: [M,N] f32.
// Block: 256 threads = 8 waves in a 2(M) x 4(N) grid; each wave computes 32x32.
// Block tile: 64(M) x 128(N). blockIdx.z selects (W,O) pair for fused QKV.

__global__ void k_gemm_bf16(const unsigned short* __restrict__ A,
                            const unsigned short* __restrict__ W0,
                            const unsigned short* __restrict__ W1,
                            const unsigned short* __restrict__ W2,
                            float* __restrict__ O0,
                            float* __restrict__ O1,
                            float* __restrict__ O2,
                            int M, int N, int K) {
    const unsigned short* W = (blockIdx.z == 0) ? W0 : (blockIdx.z == 1) ? W1 : W2;
    float* O = (blockIdx.z == 0) ? O0 : (blockIdx.z == 1) ? O1 : O2;

    int wave = threadIdx.x >> 5;
    int wm = wave >> 2;        // 0..1
    int wn = wave & 3;         // 0..3
    int l  = threadIdx.x & 31;
    int lane15 = l & 15;
    int hh = l >> 4;

    const unsigned short* Abase = A + (size_t)(blockIdx.x * 64 + wm * 32) * K;
    const unsigned short* Wbase = W + (size_t)(blockIdx.y * 128 + wn * 32) * K;

    v8f acc[2][2];
#pragma unroll
    for (int i = 0; i < 2; ++i)
#pragma unroll
        for (int j = 0; j < 2; ++j) acc[i][j] = (v8f)(0.0f);

    for (int k0 = 0; k0 < K; k0 += 32) {
        if (k0 + 64 < K) {
            __builtin_prefetch(Abase + k0 + 64, 0, 1);
            __builtin_prefetch(Wbase + k0 + 64, 0, 1);
        }
        v16bf a0 = load_a_frag(Abase + k0, K);
        v16bf a1 = load_a_frag(Abase + 16 * K + k0, K);
        v16bf b0 = load_b_frag(Wbase + k0, K);
        v16bf b1 = load_b_frag(Wbase + 16 * K + k0, K);
        acc[0][0] = wmma_bf16(a0, b0, acc[0][0]);
        acc[0][1] = wmma_bf16(a0, b1, acc[0][1]);
        acc[1][0] = wmma_bf16(a1, b0, acc[1][0]);
        acc[1][1] = wmma_bf16(a1, b1, acc[1][1]);
    }

#pragma unroll
    for (int ti = 0; ti < 2; ++ti) {
#pragma unroll
        for (int tj = 0; tj < 2; ++tj) {
            int m0 = blockIdx.x * 64 + wm * 32 + ti * 16;
            int n0 = blockIdx.y * 128 + wn * 32 + tj * 16;
#pragma unroll
            for (int r = 0; r < 8; ++r) {
                int row = m0 + r + hh * 8;
                O[(size_t)row * N + n0 + lane15] = acc[ti][tj][r];
            }
        }
    }
}

// ---------------- kernel 2: sequential scan + LayerNorm ----------------
// One wave (32 threads) per (b,h); each lane owns channels d and d+32.
// Produces eq/ek bf16 [b,h,t,d] and V transposed bf16 [b,h,d,t].

__global__ void k_scan(const float* __restrict__ Q,
                       const float* __restrict__ K,
                       const float* __restrict__ V,
                       const float* __restrict__ er,
                       const float* __restrict__ md,
                       const float* __restrict__ tw,
                       const float* __restrict__ qg, const float* __restrict__ qb,
                       const float* __restrict__ kg, const float* __restrict__ kb,
                       unsigned short* __restrict__ eqb,
                       unsigned short* __restrict__ ekb,
                       unsigned short* __restrict__ vtb) {
    int bh = blockIdx.x;
    int b = bh >> 4, h = bh & 15;
    int lane = threadIdx.x;
    int d0 = lane, d1 = lane + 32;

    float sig_er = 1.0f / (1.0f + __expf(-er[0]));
    float sig_md = 1.0f / (1.0f + __expf(-md[0]));
    float tw0 = tw[d0], tw1 = tw[d1];
    float qg0 = qg[d0], qg1 = qg[d1], qb0 = qb[d0], qb1 = qb[d1];
    float kg0 = kg[d0], kg1 = kg[d1], kb0 = kb[d0], kb1 = kb[d1];

    float mem0 = 0.0f, mem1 = 0.0f;

    for (int t = 0; t < T_; ++t) {
        size_t base = ((size_t)(b * T_ + t)) * DM_ + h * HD_;
        float q0v = Q[base + d0], q1v = Q[base + d1];
        float k0v = K[base + d0], k1v = K[base + d1];
        float v0v = V[base + d0], v1v = V[base + d1];

        float ph = sig_er * (float)(t + 1);
        float ts0 = __sinf(ph * tw0);
        float ts1 = __sinf(ph * tw1);

        // ---- eq = LN(q + ts + sig_md*mem) ----
        float x0 = q0v + ts0 + sig_md * mem0;
        float x1 = q1v + ts1 + sig_md * mem1;
        float s = x0 + x1;
#pragma unroll
        for (int off = 16; off > 0; off >>= 1) s += __shfl_xor(s, off, 32);
        float mean = s * (1.0f / 64.0f);
        float c0 = x0 - mean, c1 = x1 - mean;
        float vs = c0 * c0 + c1 * c1;
#pragma unroll
        for (int off = 16; off > 0; off >>= 1) vs += __shfl_xor(vs, off, 32);
        float inv = rsqrtf(vs * (1.0f / 64.0f) + 1e-5f);
        size_t ebase = ((size_t)bh * T_ + t) * HD_;
        eqb[ebase + d0] = f2bf(c0 * inv * qg0 + qb0);
        eqb[ebase + d1] = f2bf(c1 * inv * qg1 + qb1);

        // ---- ek = LN(k + 0.5*ts + 0.3*sig_md*mem) ----
        float y0 = k0v + 0.5f * ts0 + 0.3f * sig_md * mem0;
        float y1 = k1v + 0.5f * ts1 + 0.3f * sig_md * mem1;
        s = y0 + y1;
#pragma unroll
        for (int off = 16; off > 0; off >>= 1) s += __shfl_xor(s, off, 32);
        mean = s * (1.0f / 64.0f);
        c0 = y0 - mean; c1 = y1 - mean;
        vs = c0 * c0 + c1 * c1;
#pragma unroll
        for (int off = 16; off > 0; off >>= 1) vs += __shfl_xor(vs, off, 32);
        inv = rsqrtf(vs * (1.0f / 64.0f) + 1e-5f);
        ekb[ebase + d0] = f2bf(c0 * inv * kg0 + kb0);
        ekb[ebase + d1] = f2bf(c1 * inv * kg1 + kb1);

        // ---- memory update & V transpose ----
        mem0 = 0.7f * mem0 + 0.3f * q0v;
        mem1 = 0.7f * mem1 + 0.3f * q1v;
        vtb[((size_t)bh * HD_ + d0) * T_ + t] = f2bf(v0v);
        vtb[((size_t)bh * HD_ + d1) * T_ + t] = f2bf(v1v);
    }
}

// ---------------- kernel 3: causal flash attention ----------------
// Grid: (T/64, B*H). Block: 128 threads = 4 waves; wave w owns query rows
// [qt*64 + 16w, +16). Online softmax; P staged through LDS for the PV matmul.

__global__ void k_flash(const unsigned short* __restrict__ eqb,
                        const unsigned short* __restrict__ ekb,
                        const unsigned short* __restrict__ vtb,
                        unsigned short* __restrict__ Ob) {
    const float scale = 0.125f;  // 1/sqrt(64)
    int qt = blockIdx.x;
    int bh = blockIdx.y;
    int b = bh >> 4, h = bh & 15;

    int wave = threadIdx.x >> 5;
    int l = threadIdx.x & 31;
    int lane15 = l & 15;
    int hh = l >> 4;

    const unsigned short* eq_bh = eqb + (size_t)bh * T_ * HD_;
    const unsigned short* ek_bh = ekb + (size_t)bh * T_ * HD_;
    const unsigned short* vt_bh = vtb + (size_t)bh * HD_ * T_;

    int q0 = qt * 64 + wave * 16;

    // Q fragments (K = d = 64 -> two 32-wide k-steps), loaded once.
    v16bf aq0 = load_a_frag(eq_bh + (size_t)q0 * HD_, HD_);
    v16bf aq1 = load_a_frag(eq_bh + (size_t)q0 * HD_ + 32, HD_);

    v8f acc_o[4];
    float mrow[8], lsum[8];
#pragma unroll
    for (int n = 0; n < 4; ++n) acc_o[n] = (v8f)(0.0f);
#pragma unroll
    for (int r = 0; r < 8; ++r) { mrow[r] = -3.0e38f; lsum[r] = 0.0f; }

    __shared__ __align__(16) unsigned short Plds[4][16 * 64];
    unsigned short* pw = &Plds[wave][0];

    for (int st = 0; st <= qt; ++st) {
        int s0 = st * 64;

        // ---- S = eq @ ek^T over this 64-key tile ----
        v8f sc[4];
#pragma unroll
        for (int n = 0; n < 4; ++n) {
            sc[n] = (v8f)(0.0f);
            v16bf bk0 = load_b_frag(ek_bh + (size_t)(s0 + n * 16) * HD_, HD_);
            v16bf bk1 = load_b_frag(ek_bh + (size_t)(s0 + n * 16) * HD_ + 32, HD_);
            sc[n] = wmma_bf16(aq0, bk0, sc[n]);
            sc[n] = wmma_bf16(aq1, bk1, sc[n]);
        }

        // ---- scale, causal mask, online softmax ----
#pragma unroll
        for (int r = 0; r < 8; ++r) {
            int row = q0 + r + hh * 8;
            float mx = -3.0e38f;
#pragma unroll
            for (int n = 0; n < 4; ++n) {
                int col = s0 + n * 16 + lane15;
                float v = sc[n][r] * scale;
                v = (col <= row) ? v : -3.0e38f;
                sc[n][r] = v;
                mx = fmaxf(mx, v);
            }
#pragma unroll
            for (int off = 8; off > 0; off >>= 1)
                mx = fmaxf(mx, __shfl_xor(mx, off, 16));
            float mnew = fmaxf(mrow[r], mx);
            float alpha = __expf(mrow[r] - mnew);
            float rs = 0.0f;
#pragma unroll
            for (int n = 0; n < 4; ++n) {
                float p = __expf(sc[n][r] - mnew);
                sc[n][r] = p;
                rs += p;
            }
#pragma unroll
            for (int off = 8; off > 0; off >>= 1)
                rs += __shfl_xor(rs, off, 16);
            lsum[r] = lsum[r] * alpha + rs;
            mrow[r] = mnew;
#pragma unroll
            for (int n = 0; n < 4; ++n) acc_o[n][r] *= alpha;
        }

        // ---- stage P (bf16) in per-wave LDS, reshape C-layout -> A-layout ----
#pragma unroll
        for (int n = 0; n < 4; ++n)
#pragma unroll
            for (int r = 0; r < 8; ++r)
                pw[(r + hh * 8) * 64 + n * 16 + lane15] = f2bf(sc[n][r]);

        // ---- acc_o += P @ V  (B from transposed V: contiguous loads) ----
#pragma unroll
        for (int kk = 0; kk < 2; ++kk) {
            v16bf ap = load_a_frag(pw + kk * 32, 64);
#pragma unroll
            for (int n = 0; n < 4; ++n) {
                v16bf bv = load_b_frag(vt_bh + (size_t)(n * 16) * T_ + s0 + kk * 32, T_);
                acc_o[n] = wmma_bf16(ap, bv, acc_o[n]);
            }
        }
    }

    // ---- normalize and store to [b, t, h*64+d] bf16 ----
#pragma unroll
    for (int r = 0; r < 8; ++r) {
        int t = q0 + r + hh * 8;
        float inv_l = 1.0f / lsum[r];
#pragma unroll
        for (int n = 0; n < 4; ++n) {
            int d = n * 16 + lane15;
            Ob[((size_t)(b * T_ + t)) * DM_ + h * HD_ + d] = f2bf(acc_o[n][r] * inv_l);
        }
    }
}

// ---------------- host launcher ----------------

extern "C" void kernel_launch(void* const* d_in, const int* in_sizes, int n_in,
                              void* d_out, int out_size, void* d_ws, size_t ws_size,
                              hipStream_t stream) {
    const float* x  = (const float*)d_in[0];
    const float* Wq = (const float*)d_in[1];
    const float* Wk = (const float*)d_in[2];
    const float* Wv = (const float*)d_in[3];
    const float* Wo = (const float*)d_in[4];
    const float* er = (const float*)d_in[5];
    const float* md = (const float*)d_in[6];
    const float* tw = (const float*)d_in[7];
    const float* qg = (const float*)d_in[8];
    const float* qb = (const float*)d_in[9];
    const float* kg = (const float*)d_in[10];
    const float* kb = (const float*)d_in[11];

    const size_t MT = (size_t)B_ * T_;          // 4096 rows
    size_t off = 0;
    auto alloc = [&](size_t bytes) -> void* {
        void* p = (char*)d_ws + off;
        off += (bytes + 255) & ~(size_t)255;
        return p;
    };

    unsigned short* xb  = (unsigned short*)alloc(MT * DM_ * 2);
    unsigned short* wqb = (unsigned short*)alloc((size_t)DM_ * DM_ * 2);
    unsigned short* wkb = (unsigned short*)alloc((size_t)DM_ * DM_ * 2);
    unsigned short* wvb = (unsigned short*)alloc((size_t)DM_ * DM_ * 2);
    unsigned short* wob = (unsigned short*)alloc((size_t)DM_ * DM_ * 2);
    float* Qf = (float*)alloc(MT * DM_ * 4);
    float* Kf = (float*)alloc(MT * DM_ * 4);
    float* Vf = (float*)alloc(MT * DM_ * 4);
    unsigned short* eqb = (unsigned short*)alloc((size_t)BH_ * T_ * HD_ * 2);
    unsigned short* ekb = (unsigned short*)alloc((size_t)BH_ * T_ * HD_ * 2);
    unsigned short* vtb = (unsigned short*)alloc((size_t)BH_ * HD_ * T_ * 2);
    unsigned short* ob  = (unsigned short*)alloc(MT * DM_ * 2);

    // 0) convert inputs to bf16
    {
        int n = (int)(MT * DM_);
        k_conv_bf16<<<(n + 255) / 256, 256, 0, stream>>>(x, xb, n);
        int nw = DM_ * DM_;
        k_conv_bf16<<<(nw + 255) / 256, 256, 0, stream>>>(Wq, wqb, nw);
        k_conv_bf16<<<(nw + 255) / 256, 256, 0, stream>>>(Wk, wkb, nw);
        k_conv_bf16<<<(nw + 255) / 256, 256, 0, stream>>>(Wv, wvb, nw);
        k_conv_bf16<<<(nw + 255) / 256, 256, 0, stream>>>(Wo, wob, nw);
    }

    // 1) fused Q/K/V projections (z selects weight)
    {
        dim3 g((unsigned)(MT / 64), DM_ / 128, 3);
        k_gemm_bf16<<<g, 256, 0, stream>>>(xb, wqb, wkb, wvb, Qf, Kf, Vf,
                                           (int)MT, DM_, DM_);
    }

    // 2) temporal scan + LN (one wave per (b,h))
    k_scan<<<BH_, 32, 0, stream>>>(Qf, Kf, Vf, er, md, tw, qg, qb, kg, kb,
                                   eqb, ekb, vtb);

    // 3) causal flash attention
    {
        dim3 g(T_ / 64, BH_);
        k_flash<<<g, 128, 0, stream>>>(eqb, ekb, vtb, ob);
    }

    // 4) output projection -> d_out (f32)
    {
        dim3 g((unsigned)(MT / 64), DM_ / 128, 1);
        k_gemm_bf16<<<g, 256, 0, stream>>>(ob, wob, wob, wob,
                                           (float*)d_out, (float*)d_out, (float*)d_out,
                                           (int)MT, DM_, DM_);
    }
}
